// GatedGNN_67199058313551
// MI455X (gfx1250) — matrix operand order repaired and checked by reading, not compiled, not run on hardware
//
#include <hip/hip_runtime.h>
#include <hip/hip_bf16.h>
#include <stdint.h>

// ---------------- problem constants (from reference) ----------------
#define B_GRAPH 64
#define N_PER   512
#define C_DIM   256
#define H3      768                 // 3*C
#define NC      (B_GRAPH * N_PER)   // 32768 total node rows
#define CAP     64                  // max edges per (step, graph)
#define G_PER_WG 4                  // graphs per workgroup (16 WGs total)

typedef __attribute__((ext_vector_type(16))) __bf16        v16bf;
typedef __attribute__((ext_vector_type(8)))  float         v8f;
typedef __attribute__((ext_vector_type(4)))  unsigned int  u32x4;

union BFrag {
    v16bf v;
    unsigned short u[16];
    u32x4 q[2];
};

__device__ __forceinline__ unsigned short f2bf(float f) {
    uint32_t u = __float_as_uint(f);
    uint32_t r = u + 0x7FFFu + ((u >> 16) & 1u);   // round to nearest even
    return (unsigned short)(r >> 16);
}
__device__ __forceinline__ float bf2f(unsigned short h) {
    return __uint_as_float(((uint32_t)h) << 16);
}

// A-fragment (16x32 bf16) from a row-major bf16 [16][256] tile in LDS.
// CDNA5 layout: lanes 0-15 -> rows 0-15, K = {0..7,16..23}; lanes 16-31 same rows, K = {8..15,24..31}.
// Two contiguous 16B ds_load_b128 per lane, no conversion VALU.
__device__ __forceinline__ v16bf load_a_frag_bf(const unsigned short* base, int kt, int half, int ml) {
    BFrag f;
    const unsigned short* p = base + ml * C_DIM + kt * 32 + half * 8;
    f.q[0] = *(const u32x4*)(p);
    f.q[1] = *(const u32x4*)(p + 16);
    return f.v;
}

// B-fragment (32x16 bf16) for D = A @ W^T: W is (N,K) row-major bf16, so B[k][n] = W[n][k].
// Lane col n = lane%16; lanes 0-15 hold K=kt*32+{0..7,16..23}, lanes 16-31 hold K=kt*32+{8..15,24..31}.
__device__ __forceinline__ v16bf load_b_frag(const unsigned short* w, int ldk, int n, int kt, int half) {
    BFrag f;
    const u32x4* p = (const u32x4*)(w + (size_t)n * ldk + kt * 32 + half * 8);
    f.q[0] = p[0];
    f.q[1] = p[2];
    return f.v;
}

// ---------------- utility kernels ----------------
__global__ void copy_f32_kernel(const float* __restrict__ src, float* __restrict__ dst, int n) {
    for (int i = blockIdx.x * blockDim.x + threadIdx.x; i < n; i += gridDim.x * blockDim.x)
        dst[i] = src[i];
}

__global__ void convert_weights_kernel(const float* __restrict__ W1, const float* __restrict__ W2,
                                       const float* __restrict__ wih, const float* __restrict__ whh,
                                       unsigned short* __restrict__ w1b, unsigned short* __restrict__ w2b,
                                       unsigned short* __restrict__ wihb, unsigned short* __restrict__ whhb) {
    int i = blockIdx.x * blockDim.x + threadIdx.x;
    if (i < C_DIM * C_DIM) { w1b[i] = f2bf(W1[i]); w2b[i] = f2bf(W2[i]); }
    if (i < H3 * C_DIM)    { wihb[i] = f2bf(wih[i]); whhb[i] = f2bf(whh[i]); }
}

// Compact the padded (N_PER, K) edge rows into per-(step, graph) lists.
__global__ void build_edges_kernel(const int* __restrict__ src, const int* __restrict__ gid,
                                   const float* __restrict__ msk, int K,
                                   int* __restrict__ ecnt, int* __restrict__ esrc) {
    int idx = blockIdx.x * blockDim.x + threadIdx.x;   // step*64 + graph
    if (idx >= N_PER * B_GRAPH) return;
    int step = idx >> 6;
    int g    = idx & 63;
    int cnt  = 0;
    for (int k = 0; k < K; ++k) {
        int   gg = gid[step * K + k];
        float mm = msk[step * K + k];
        if (gg == g && mm != 0.0f) {
            if (cnt < CAP) esrc[(size_t)idx * CAP + cnt] = src[step * K + k];
            ++cnt;
        }
    }
    ecnt[idx] = cnt < CAP ? cnt : CAP;
}

// gh = hx @ w_hh^T + b_hh  for all 32768 rows, output bf16.  grid = (2048, 6), 256 threads.
__global__ __launch_bounds__(256) void gh_kernel(const float* __restrict__ hx,
                                                 const unsigned short* __restrict__ whhb,
                                                 const float* __restrict__ b_hh,
                                                 unsigned short* __restrict__ gh) {
    __shared__ unsigned short atb[16][C_DIM];   // bf16 A tile
    const int tid = threadIdx.x;
    const int wave = tid >> 5, lane = tid & 31, half = (lane >> 4) & 1, ml = lane & 15;
    const int mtile = blockIdx.x;             // 16-row tile of hx
    const int nbase = blockIdx.y * 128;       // 128-col slab, one 16-col tile per wave

    for (int i = tid; i < 16 * C_DIM; i += 256) {
        int m = i >> 8, k = i & 255;
        atb[m][k] = f2bf(hx[((size_t)mtile * 16 + m) * C_DIM + k]);   // convert once
    }
    __syncthreads();

    const int n = nbase + wave * 16 + ml;
    v8f acc = {0.f, 0.f, 0.f, 0.f, 0.f, 0.f, 0.f, 0.f};
#pragma unroll
    for (int kt = 0; kt < 8; ++kt) {
        v16bf A  = load_a_frag_bf(&atb[0][0], kt, half, ml);
        v16bf Bv = load_b_frag(whhb, C_DIM, n, kt, half);
        acc = __builtin_amdgcn_wmma_f32_16x16x32_bf16(false, A, false, Bv, (short)0, acc, false, false);
    }
    const float bb = b_hh[n];
#pragma unroll
    for (int v = 0; v < 8; ++v) {
        int m = v + half * 8;
        gh[((size_t)mtile * 16 + m) * H3 + n] = f2bf(acc[v] + bb);
    }
}

// ---------------- persistent scan kernel ----------------
// 16 workgroups x 256 threads (8 wave32). Each WG owns 4 graphs and runs the full
// 512-step serial scan for one direction. No cross-WG sync needed (graphs independent).
__global__ __launch_bounds__(256) void scan_kernel(float* __restrict__ hx,
                                                   unsigned short* __restrict__ y,
                                                   const unsigned short* __restrict__ gh,
                                                   const unsigned short* __restrict__ wihb,
                                                   const unsigned short* __restrict__ w1b,
                                                   const unsigned short* __restrict__ w2b,
                                                   const float* __restrict__ b_ih,
                                                   const float* __restrict__ b1,
                                                   const float* __restrict__ b2,
                                                   const int* __restrict__ ecnt,
                                                   const int* __restrict__ esrc,
                                                   int dir) {
    __shared__ float          aggf [G_PER_WG][C_DIM];   // f32 gather accumulators (ds_add_f32)
    __shared__ unsigned short aggb [16][C_DIM];         // bf16 A tile (rows >= 4 zero)
    __shared__ float          gi   [G_PER_WG][H3];
    __shared__ unsigned short hnewb[16][C_DIM];         // bf16 A tile (rows >= 4 zero)
    __shared__ unsigned short amidb[16][C_DIM];         // bf16 A tile (rows >= 4 zero)
    __shared__ float          bih_s[H3];
    __shared__ float          b1_s [C_DIM];
    __shared__ float          b2_s [C_DIM];

    const int tid  = threadIdx.x;
    const int wave = tid >> 5, lane = tid & 31, half = (lane >> 4) & 1, ml = lane & 15;
    const int g0   = blockIdx.x * G_PER_WG;

    // one-time init: cache biases in LDS, zero the padded rows of bf16 tiles
    for (int i = tid; i < H3; i += 256) bih_s[i] = b_ih[i];
    if (tid < C_DIM) { b1_s[tid] = b1[tid]; b2_s[tid] = b2[tid]; }
    for (int i = tid; i < 16 * C_DIM; i += 256) {
        if ((i >> 8) >= G_PER_WG) {
            (&aggb[0][0])[i]  = 0;
            (&hnewb[0][0])[i] = 0;
            (&amidb[0][0])[i] = 0;
        }
    }
    __syncthreads();

    for (int step = 0; step < N_PER; ++step) {
        const int node = dir ? (N_PER - 1 - step) : step;

        // ---- phase 0: clear gather accumulators ----
        for (int i = tid; i < G_PER_WG * C_DIM; i += 256) (&aggf[0][0])[i] = 0.f;
        __syncthreads();

        // ---- phase 1: gather  aggf[g] += y[src]  (LDS float atomics = ds_add_f32) ----
        {
            const int gl    = tid >> 6;          // 0..3 local graph
            const int l     = tid & 63;
            const int sslot = l >> 4;            // 4-way parallel over edges
            const int cb    = (l & 15) * 16;     // 16 channels per thread
            const int eidx  = step * B_GRAPH + g0 + gl;
            const int cnt   = ecnt[eidx];
            for (int s = sslot; s < cnt; s += 4) {
                const int srow = esrc[(size_t)eidx * CAP + s];
                const unsigned short* yr = y + (size_t)srow * C_DIM + cb;
                u32x4 q0 = *(const u32x4*)(yr);
                u32x4 q1 = *(const u32x4*)(yr + 8);
                const unsigned short* u0 = (const unsigned short*)&q0;
                const unsigned short* u1 = (const unsigned short*)&q1;
#pragma unroll
                for (int cc = 0; cc < 8; ++cc) {
                    atomicAdd(&aggf[gl][cb + cc],     bf2f(u0[cc]));
                    atomicAdd(&aggf[gl][cb + 8 + cc], bf2f(u1[cc]));
                }
            }
        }
        __syncthreads();

        // ---- phase 1.5: convert gather result to bf16 A tile (once per value) ----
        {
            const int c = tid;
#pragma unroll
            for (int m = 0; m < G_PER_WG; ++m) aggb[m][c] = f2bf(aggf[m][c]);
        }
        __syncthreads();

        // ---- phase 2: GRU input GEMM  gi = agg @ w_ih^T   (M=16 pad, N=768, K=256) ----
        {
            v16bf afr[8];
#pragma unroll
            for (int kt = 0; kt < 8; ++kt) afr[kt] = load_a_frag_bf(&aggb[0][0], kt, half, ml);
            for (int t = 0; t < 6; ++t) {
                const int nt = wave * 6 + t;
                const int n  = nt * 16 + ml;
                v8f acc = {0.f, 0.f, 0.f, 0.f, 0.f, 0.f, 0.f, 0.f};
#pragma unroll
                for (int kt = 0; kt < 8; ++kt) {
                    v16bf Bv = load_b_frag(wihb, C_DIM, n, kt, half);
                    acc = __builtin_amdgcn_wmma_f32_16x16x32_bf16(false, afr[kt], false, Bv, (short)0, acc, false, false);
                }
                if (half == 0) {
#pragma unroll
                    for (int v = 0; v < G_PER_WG; ++v) gi[v][n] = acc[v];
                }
            }
        }
        __syncthreads();

        // ---- phase 3: gates (VALU) + hx update ----
        {
            const int c = tid;   // one channel per thread
#pragma unroll
            for (int m = 0; m < G_PER_WG; ++m) {
                const size_t row = (size_t)(g0 + m) * N_PER + node;
                const float ir  = gi[m][c]             + bih_s[c];
                const float iz  = gi[m][C_DIM + c]     + bih_s[C_DIM + c];
                const float in_ = gi[m][2 * C_DIM + c] + bih_s[2 * C_DIM + c];
                const unsigned short* ghr = gh + row * H3;
                const float hr = bf2f(ghr[c]);
                const float hz = bf2f(ghr[C_DIM + c]);
                const float hn = bf2f(ghr[2 * C_DIM + c]);
                const float hold = hx[row * C_DIM + c];
                const float r  = 1.f / (1.f + __expf(-(ir + hr)));
                const float z  = 1.f / (1.f + __expf(-(iz + hz)));
                const float nn = tanhf(in_ + r * hn);
                const float hv = (1.f - z) * nn + z * hold;
                hnewb[m][c] = f2bf(hv);        // bf16 for MLP A tile
                hx[row * C_DIM + c] = hv;      // f32 state
            }
        }
        __syncthreads();

        // ---- phase 4: MLP layer 1  amid = relu(hnew @ W1^T + b1) ----
        {
            v16bf afr[8];
#pragma unroll
            for (int kt = 0; kt < 8; ++kt) afr[kt] = load_a_frag_bf(&hnewb[0][0], kt, half, ml);
            for (int t = 0; t < 2; ++t) {
                const int nt = wave * 2 + t;
                const int n  = nt * 16 + ml;
                v8f acc = {0.f, 0.f, 0.f, 0.f, 0.f, 0.f, 0.f, 0.f};
#pragma unroll
                for (int kt = 0; kt < 8; ++kt) {
                    v16bf Bv = load_b_frag(w1b, C_DIM, n, kt, half);
                    acc = __builtin_amdgcn_wmma_f32_16x16x32_bf16(false, afr[kt], false, Bv, (short)0, acc, false, false);
                }
                if (half == 0) {
                    const float bb = b1_s[n];
#pragma unroll
                    for (int v = 0; v < G_PER_WG; ++v) {
                        float xv = acc[v] + bb;
                        amidb[v][n] = f2bf(xv > 0.f ? xv : 0.f);
                    }
                }
            }
        }
        __syncthreads();

        // ---- phase 5: MLP layer 2  y[node rows] = relu(amid @ W2^T + b2) (bf16) ----
        {
            v16bf afr[8];
#pragma unroll
            for (int kt = 0; kt < 8; ++kt) afr[kt] = load_a_frag_bf(&amidb[0][0], kt, half, ml);
            for (int t = 0; t < 2; ++t) {
                const int nt = wave * 2 + t;
                const int n  = nt * 16 + ml;
                v8f acc = {0.f, 0.f, 0.f, 0.f, 0.f, 0.f, 0.f, 0.f};
#pragma unroll
                for (int kt = 0; kt < 8; ++kt) {
                    v16bf Bv = load_b_frag(w2b, C_DIM, n, kt, half);
                    acc = __builtin_amdgcn_wmma_f32_16x16x32_bf16(false, afr[kt], false, Bv, (short)0, acc, false, false);
                }
                if (half == 0) {
                    const float bb = b2_s[n];
#pragma unroll
                    for (int v = 0; v < G_PER_WG; ++v) {
                        float xv = acc[v] + bb;
                        xv = xv > 0.f ? xv : 0.f;
                        const size_t row = (size_t)(g0 + v) * N_PER + node;
                        y[row * C_DIM + n] = f2bf(xv);
                    }
                }
            }
        }
        __threadfence();   // make y / hx stores visible to our own next-step loads
        __syncthreads();
    }
}

// ---------------- host launch ----------------
extern "C" void kernel_launch(void* const* d_in, const int* in_sizes, int n_in,
                              void* d_out, int out_size, void* d_ws, size_t ws_size,
                              hipStream_t stream) {
    const float* x    = (const float*)d_in[0];
    const float* W1   = (const float*)d_in[1];
    const float* b1   = (const float*)d_in[2];
    const float* W2   = (const float*)d_in[3];
    const float* b2   = (const float*)d_in[4];
    const float* w_ih = (const float*)d_in[5];
    const float* b_ih = (const float*)d_in[6];
    const float* w_hh = (const float*)d_in[7];
    const float* b_hh = (const float*)d_in[8];
    const int*   src_f = (const int*)d_in[9];
    const int*   gid_f = (const int*)d_in[10];
    const float* msk_f = (const float*)d_in[11];
    const int*   src_b = (const int*)d_in[12];
    const int*   gid_b = (const int*)d_in[13];
    const float* msk_b = (const float*)d_in[14];

    const int Kf = in_sizes[9]  / N_PER;
    const int Kb = in_sizes[12] / N_PER;

    float* hx = (float*)d_out;   // 32768 x 256 f32 working state == final output

    // ---- workspace carve ----
    char* p = (char*)d_ws;
    auto carve = [&](size_t bytes) {
        char* r = p;
        p += (bytes + 255) & ~(size_t)255;
        return r;
    };
    unsigned short* w1b  = (unsigned short*)carve((size_t)C_DIM * C_DIM * 2);
    unsigned short* w2b  = (unsigned short*)carve((size_t)C_DIM * C_DIM * 2);
    unsigned short* wihb = (unsigned short*)carve((size_t)H3 * C_DIM * 2);
    unsigned short* whhb = (unsigned short*)carve((size_t)H3 * C_DIM * 2);
    unsigned short* yb   = (unsigned short*)carve((size_t)NC * C_DIM * 2);   // 16 MB
    unsigned short* ghb  = (unsigned short*)carve((size_t)NC * H3 * 2);      // 48 MB
    int* ecnt_f = (int*)carve((size_t)N_PER * B_GRAPH * 4);
    int* esrc_f = (int*)carve((size_t)N_PER * B_GRAPH * CAP * 4);            // 8 MB
    int* ecnt_b = (int*)carve((size_t)N_PER * B_GRAPH * 4);
    int* esrc_b = (int*)carve((size_t)N_PER * B_GRAPH * CAP * 4);            // 8 MB
    (void)ws_size; (void)n_in; (void)out_size;

    // 1) hx = x
    copy_f32_kernel<<<4096, 256, 0, stream>>>(x, hx, NC * C_DIM);
    // 2) weights -> bf16
    convert_weights_kernel<<<(H3 * C_DIM + 255) / 256, 256, 0, stream>>>(
        W1, W2, w_ih, w_hh, w1b, w2b, wihb, whhb);
    // 3) compact edge lists (indexed by scan step)
    build_edges_kernel<<<(N_PER * B_GRAPH + 255) / 256, 256, 0, stream>>>(src_f, gid_f, msk_f, Kf, ecnt_f, esrc_f);
    build_edges_kernel<<<(N_PER * B_GRAPH + 255) / 256, 256, 0, stream>>>(src_b, gid_b, msk_b, Kb, ecnt_b, esrc_b);
    // 4) gh for forward scan, then the forward scan
    gh_kernel<<<dim3(NC / 16, H3 / 128), 256, 0, stream>>>(hx, whhb, b_hh, ghb);
    scan_kernel<<<B_GRAPH / G_PER_WG, 256, 0, stream>>>(hx, yb, ghb, wihb, w1b, w2b,
                                                        b_ih, b1, b2, ecnt_f, esrc_f, 0);
    // 5) gh for backward scan (hx now updated), then the backward scan
    gh_kernel<<<dim3(NC / 16, H3 / 128), 256, 0, stream>>>(hx, whhb, b_hh, ghb);
    scan_kernel<<<B_GRAPH / G_PER_WG, 256, 0, stream>>>(hx, yb, ghb, wihb, w1b, w2b,
                                                        b_ih, b1, b2, ecnt_b, esrc_b, 1);
}
